// SimpleTransformerBlock_29953101922405
// MI455X (gfx1250) — compile-verified, matrix operand order checked by reading, not run on hardware
//
#include <hip/hip_runtime.h>

#define DIMD   1024
#define SEQ    2048
#define BATCH  2
#define NHEAD  16
#define HD     64
#define FFDIM  4096

typedef __attribute__((ext_vector_type(16))) __bf16       v16bf;
typedef __attribute__((ext_vector_type(8)))  float        v8f;
typedef __attribute__((ext_vector_type(8)))  unsigned int v8u;
typedef __attribute__((ext_vector_type(4)))  unsigned int u32x4;
typedef __attribute__((ext_vector_type(8)))  unsigned int u32x8;

union FragU { v8u u; v16bf b; };
union U4S  { uint4 v; unsigned short s[8]; };

__device__ __forceinline__ unsigned short f2bf(float f) {
  unsigned int u = __float_as_uint(f);
  u += 0x7fffu + ((u >> 16) & 1u);           // round-to-nearest-even
  return (unsigned short)(u >> 16);
}

// Async global->LDS 16-byte copy (ASYNCcnt path, no VGPR round-trip).
__device__ __forceinline__ void async_copy_b128(const void* lds_dst, const void* gsrc) {
  unsigned loff = (unsigned)(unsigned long long)lds_dst;   // LDS byte offset
  asm volatile("global_load_async_to_lds_b128 %0, %1, off"
               :: "v"(loff), "v"(gsrc) : "memory");
}
__device__ __forceinline__ void wait_async0() {
  asm volatile("s_wait_asynccnt 0x0" ::: "memory");
}

// Tensor Data Mover: 2D tile of bf16 (tile_dim0 x tile_dim1 elements) from a
// row-major tensor with row stride `stride_elems` into LDS at `lds_dst`,
// inserting 4 dwords of padding after every 32 dwords (-> 36-dword row pitch
// for a 64-element row, matching frag_ld's conflict-free layout).
__device__ __forceinline__ void tdm_load_tile_bf16(const void* lds_dst, const void* gsrc,
                                                   unsigned tile_d0, unsigned tile_d1,
                                                   unsigned tensor_d0, unsigned tensor_d1,
                                                   unsigned stride_elems) {
  unsigned long long ga = (unsigned long long)gsrc;
  u32x4 g0;
  g0[0] = 1u;                                            // count=1, is_restore=0
  g0[1] = (unsigned)(unsigned long long)lds_dst;         // lds_addr (bytes)
  g0[2] = (unsigned)ga;                                  // global_addr[31:0]
  g0[3] = (unsigned)(ga >> 32) | (2u << 30);             // global_addr[56:32] | type=2
  u32x8 g1;
  g1[0] = (1u << 16)                                     // data_size = 2 bytes
        | (1u << 20)                                     // pad_enable
        | (4u << 22)                                     // pad_interval: 32 dwords
        | (3u << 25);                                    // pad_amount: 4 dwords
  g1[1] = (tensor_d0 & 0xffffu) << 16;                   // tensor_dim0[15:0]
  g1[2] = (tensor_d0 >> 16) | ((tensor_d1 & 0xffffu) << 16);
  g1[3] = (tensor_d1 >> 16) | (tile_d0 << 16);           // tile_dim0
  g1[4] = tile_d1;                                       // tile_dim1 (tile_dim2=0)
  g1[5] = stride_elems;                                  // tensor_dim0_stride[31:0]
  g1[6] = 0u;
  g1[7] = 0u;
  asm volatile("tensor_load_to_lds %0, %1" :: "s"(g0), "s"(g1) : "memory");
}

// Load a 16x32 bf16 WMMA operand (A-type or B-type) from LDS stored as packed
// bf16 pairs: base[row * ldu + kpair].  ISA layout: lanes 0-15 hold K{0..7,16..23},
// lanes 16-31 hold K{8..15,24..31}; pairs contiguous within each VGPR.
__device__ __forceinline__ v16bf frag_ld(const unsigned int* base, int row0, int ldu, int kp0) {
  const int lane = threadIdx.x & 31;
  const int half = (lane >> 4) & 1;
  const unsigned int* p = base + (row0 + (lane & 15)) * ldu + kp0 + half * 4;
  uint4 lo = *(const uint4*)(p);       // K pairs 0..3 / 4..7
  uint4 hi = *(const uint4*)(p + 8);   // K pairs 8..11 / 12..15
  FragU f;
  f.u[0] = lo.x; f.u[1] = lo.y; f.u[2] = lo.z; f.u[3] = lo.w;
  f.u[4] = hi.x; f.u[5] = hi.y; f.u[6] = hi.z; f.u[7] = hi.w;
  return f.b;
}

__device__ __forceinline__ v8f wmma_bf16(v16bf a, v16bf b, v8f c) {
  return __builtin_amdgcn_wmma_f32_16x16x32_bf16(false, a, false, b, (short)0, c, false, false);
}

// ---------------------------------------------------------------- converts
__global__ void __launch_bounds__(256) k_f2bf(const float* __restrict__ in,
                                              unsigned short* __restrict__ out, int n) {
  for (int i = blockIdx.x * 256 + threadIdx.x; i < n; i += gridDim.x * 256)
    out[i] = f2bf(in[i]);
}

// ---------------------------------------------------------------- layernorm
__global__ void __launch_bounds__(256) k_layernorm(const float* __restrict__ x,
                                                   const float* __restrict__ g,
                                                   const float* __restrict__ bt,
                                                   unsigned short* __restrict__ out) {
  const int row = blockIdx.x, tid = threadIdx.x;
  const int lane = tid & 31, wave = tid >> 5;
  const float4 xv = *(const float4*)(x + (size_t)row * DIMD + tid * 4);
  float s = xv.x + xv.y + xv.z + xv.w;
  float q = xv.x * xv.x + xv.y * xv.y + xv.z * xv.z + xv.w * xv.w;
#pragma unroll
  for (int m = 16; m; m >>= 1) { s += __shfl_xor(s, m, 32); q += __shfl_xor(q, m, 32); }
  __shared__ float ss[8], sq[8];
  if (lane == 0) { ss[wave] = s; sq[wave] = q; }
  __syncthreads();
  float ts = 0.0f, tq = 0.0f;
#pragma unroll
  for (int i = 0; i < 8; ++i) { ts += ss[i]; tq += sq[i]; }
  const float mu   = ts * (1.0f / DIMD);
  const float var  = tq * (1.0f / DIMD) - mu * mu;
  const float rstd = rsqrtf(var + 1e-5f);
  const float4 gv = *(const float4*)(g + tid * 4);
  const float4 bv = *(const float4*)(bt + tid * 4);
  unsigned int p0 = (unsigned int)f2bf((xv.x - mu) * rstd * gv.x + bv.x)
                  | ((unsigned int)f2bf((xv.y - mu) * rstd * gv.y + bv.y) << 16);
  unsigned int p1 = (unsigned int)f2bf((xv.z - mu) * rstd * gv.z + bv.z)
                  | ((unsigned int)f2bf((xv.w - mu) * rstd * gv.w + bv.w) << 16);
  uint2 pk; pk.x = p0; pk.y = p1;
  *(uint2*)(out + (size_t)row * DIMD + tid * 4) = pk;
}

// ---------------------------------------------------------------- bf16 GEMM
// C[M,N] = A[M,K](bf16) * W[K,N](bf16) + bias (+ gelu) (+ residual)
// block tile 128x128, 8 waves in 2(M)x4(N), wave tile 64x32, K-step 64.
// A tile is fetched by the Tensor Data Mover (one descriptor per K step,
// issued by wave 0); TDM padding reproduces the 36-dword LDS row pitch.
__global__ void __launch_bounds__(256) k_gemm_bf16(const unsigned short* __restrict__ A,
                                                   const unsigned short* __restrict__ W,
                                                   const float* __restrict__ bias,
                                                   const float* __restrict__ residual,
                                                   float* __restrict__ outF,
                                                   unsigned short* __restrict__ outB,
                                                   int M, int N, int K, int gelu) {
  __shared__ unsigned int lds_a[128 * 36];   // [row][kpair], 36-dword pitch (TDM pad)
  __shared__ unsigned int lds_w[128 * 36];   // [col][kpair] (transposed-packed)
  const int tid = threadIdx.x, lane = tid & 31, wave = tid >> 5;
  const int wm = wave >> 2, wn = wave & 3;
  const int m0 = blockIdx.y * 128, n0 = blockIdx.x * 128;
  v8f acc[4][2] = {};
  for (int k0 = 0; k0 < K; k0 += 64) {
    __syncthreads();
    if (wave == 0)                            // A tile 128x64 bf16 via TDM
      tdm_load_tile_bf16(&lds_a[0], A + (size_t)m0 * K + k0,
                         64u, 128u, (unsigned)K, (unsigned)M, (unsigned)K);
#pragma unroll
    for (int i = 0; i < 2; ++i) {             // W tile 64x128 -> transposed pairs
      int c = tid + i * 256;
      int kp = c >> 4, nc = c & 15;
      U4S r0, r1;
      r0.v = *(const uint4*)(W + (size_t)(k0 + 2 * kp)     * N + n0 + nc * 8);
      r1.v = *(const uint4*)(W + (size_t)(k0 + 2 * kp + 1) * N + n0 + nc * 8);
      if (k0 + 64 < K)
        __builtin_prefetch(W + (size_t)(k0 + 64 + 2 * kp) * N + n0 + nc * 8, 0, 1);
#pragma unroll
      for (int j = 0; j < 8; ++j)
        lds_w[(nc * 8 + j) * 36 + kp] = (unsigned int)r0.s[j] | ((unsigned int)r1.s[j] << 16);
    }
    if (wave == 0) __builtin_amdgcn_s_wait_tensorcnt(0);
    __syncthreads();
#pragma unroll
    for (int kk = 0; kk < 2; ++kk) {
      v16bf aF[4], bF[2];
#pragma unroll
      for (int mt = 0; mt < 4; ++mt) aF[mt] = frag_ld(lds_a, wm * 64 + mt * 16, 36, kk * 16);
#pragma unroll
      for (int nt = 0; nt < 2; ++nt) bF[nt] = frag_ld(lds_w, wn * 32 + nt * 16, 36, kk * 16);
#pragma unroll
      for (int mt = 0; mt < 4; ++mt)
#pragma unroll
        for (int nt = 0; nt < 2; ++nt)
          acc[mt][nt] = wmma_bf16(aF[mt], bF[nt], acc[mt][nt]);
    }
  }
  const int halfr = lane >> 4, nl = lane & 15;
#pragma unroll
  for (int mt = 0; mt < 4; ++mt)
#pragma unroll
    for (int nt = 0; nt < 2; ++nt)
#pragma unroll
      for (int r = 0; r < 8; ++r) {
        int m = m0 + wm * 64 + mt * 16 + halfr * 8 + r;
        int n = n0 + wn * 32 + nt * 16 + nl;
        float v = acc[mt][nt][r] + bias[n];
        if (gelu) v = 0.5f * v * (1.0f + erff(v * 0.70710678118654752f));
        if (residual) v += residual[(size_t)m * N + n];
        if (outF) outF[(size_t)m * N + n] = v;
        else      outB[(size_t)m * N + n] = f2bf(v);
      }
}

// ---------------------------------------------------------------- attention
// Flash-style: block = 128 query rows of one (b,h); wave owns 16 rows.
// Q/K tiles stream via async global->LDS; V is transpose-packed via VGPRs.
__global__ void __launch_bounds__(256) k_attention(const unsigned short* __restrict__ Q,
                                                   const unsigned short* __restrict__ Kt,
                                                   const unsigned short* __restrict__ V,
                                                   unsigned short* __restrict__ O) {
  __shared__ unsigned int q_lds[128 * 36];   // [q row][hd pair]
  __shared__ unsigned int k_lds[64 * 36];    // [key][hd pair]
  __shared__ unsigned int v_lds[64 * 36];    // [hd col][key pair] (transposed-packed)
  __shared__ unsigned int p_lds[8 * 16 * 36];// per-wave P scratch [row][key pair]
  const int tid = threadIdx.x, lane = tid & 31, wave = tid >> 5;
  const int q0 = blockIdx.x * 128;
  const int bh = blockIdx.y;
  const size_t base = ((size_t)(bh >> 4) * SEQ) * DIMD + (size_t)(bh & 15) * HD;
#pragma unroll
  for (int i = 0; i < 4; ++i) {              // Q tile 128x64 (async)
    int c = tid + i * 256;
    int row = c >> 3, cc = c & 7;
    async_copy_b128(&q_lds[row * 36 + cc * 4],
                    Q + base + (size_t)(q0 + row) * DIMD + cc * 8);
  }
  v8f o[4] = {};
  float mrow[8], lrow[8];
#pragma unroll
  for (int r = 0; r < 8; ++r) { mrow[r] = -3.0e38f; lrow[r] = 0.0f; }
  const float scale = 0.125f;                // 1/sqrt(64)
  unsigned short* pw = (unsigned short*)&p_lds[wave * 16 * 36];
  for (int kb = 0; kb < SEQ / 64; ++kb) {
    const int kbase = kb * 64;
    __syncthreads();                         // prior readers of k_lds/v_lds done
#pragma unroll
    for (int i = 0; i < 2; ++i) {            // K tile 64x64 (async)
      int c = tid + i * 256;
      int row = c >> 3, cc = c & 7;
      async_copy_b128(&k_lds[row * 36 + cc * 4],
                      Kt + base + (size_t)(kbase + row) * DIMD + cc * 8);
    }
    {                                        // V tile 64x64 transposed-packed
      int kp = tid >> 3, cc = tid & 7;
      U4S r0, r1;
      r0.v = *(const uint4*)(V + base + (size_t)(kbase + 2 * kp)     * DIMD + cc * 8);
      r1.v = *(const uint4*)(V + base + (size_t)(kbase + 2 * kp + 1) * DIMD + cc * 8);
#pragma unroll
      for (int j = 0; j < 8; ++j)
        v_lds[(cc * 8 + j) * 36 + kp] = (unsigned int)r0.s[j] | ((unsigned int)r1.s[j] << 16);
    }
    wait_async0();                           // own async K (+Q first iter) done
    __syncthreads();
    v8f st[4] = {};                          // S tile: 16 q x 64 keys
#pragma unroll
    for (int kk = 0; kk < 2; ++kk) {
      v16bf a = frag_ld(q_lds, wave * 16, 36, kk * 16);
#pragma unroll
      for (int t = 0; t < 4; ++t) {
        v16bf bf = frag_ld(k_lds, t * 16, 36, kk * 16);
        st[t] = wmma_bf16(a, bf, st[t]);
      }
    }
    const int rowoff = (lane >> 4) * 8;
#pragma unroll
    for (int r = 0; r < 8; ++r) {            // online softmax per row
      float mx = fmaxf(fmaxf(st[0][r], st[1][r]), fmaxf(st[2][r], st[3][r]));
#pragma unroll
      for (int m = 1; m <= 8; m <<= 1) mx = fmaxf(mx, __shfl_xor(mx, m, 32));
      const float mnew  = fmaxf(mrow[r], mx * scale);
      const float alpha = __expf(mrow[r] - mnew);
      float sum = 0.0f;
#pragma unroll
      for (int t = 0; t < 4; ++t) {
        float p = __expf(st[t][r] * scale - mnew);
        sum += p;
        pw[(rowoff + r) * 72 + t * 16 + (lane & 15)] = f2bf(p);
      }
#pragma unroll
      for (int m = 1; m <= 8; m <<= 1) sum += __shfl_xor(sum, m, 32);
      lrow[r] = lrow[r] * alpha + sum;
      mrow[r] = mnew;
#pragma unroll
      for (int nt = 0; nt < 4; ++nt) o[nt][r] = o[nt][r] * alpha;
    }
    // P @ V (wave-private P scratch; DS ops in-order within a wave)
#pragma unroll
    for (int kk = 0; kk < 2; ++kk) {
      v16bf a = frag_ld(&p_lds[wave * 16 * 36], 0, 36, kk * 16);
#pragma unroll
      for (int nt = 0; nt < 4; ++nt) {
        v16bf bf = frag_ld(v_lds, nt * 16, 36, kk * 16);
        o[nt] = wmma_bf16(a, bf, o[nt]);
      }
    }
  }
  const int rowoff = (lane >> 4) * 8, nl = lane & 15;
#pragma unroll
  for (int r = 0; r < 8; ++r) {
    const float inv = 1.0f / lrow[r];
#pragma unroll
    for (int nt = 0; nt < 4; ++nt) {
      int row = q0 + wave * 16 + rowoff + r;
      O[base + (size_t)row * DIMD + nt * 16 + nl] = f2bf(o[nt][r] * inv);
    }
  }
}

// ---------------------------------------------------------------- host
extern "C" void kernel_launch(void* const* d_in, const int* in_sizes, int n_in,
                              void* d_out, int out_size, void* d_ws, size_t ws_size,
                              hipStream_t stream) {
  (void)in_sizes; (void)n_in; (void)out_size; (void)ws_size;
  const float* x   = (const float*)d_in[0];
  const float* wq  = (const float*)d_in[1];  const float* bq  = (const float*)d_in[2];
  const float* wk  = (const float*)d_in[3];  const float* bk  = (const float*)d_in[4];
  const float* wv  = (const float*)d_in[5];  const float* bv  = (const float*)d_in[6];
  const float* wo  = (const float*)d_in[7];  const float* bo  = (const float*)d_in[8];
  const float* w1  = (const float*)d_in[9];  const float* b1  = (const float*)d_in[10];
  const float* w2  = (const float*)d_in[11]; const float* b2  = (const float*)d_in[12];
  const float* g1  = (const float*)d_in[13]; const float* be1 = (const float*)d_in[14];
  const float* g2  = (const float*)d_in[15]; const float* be2 = (const float*)d_in[16];

  const int M = BATCH * SEQ;                 // 4096 rows
  char* ws = (char*)d_ws;
  size_t off = 0;
  auto alloc = [&](size_t bytes) -> void* {
    void* p = (void*)(ws + off);
    off = (off + bytes + 255) & ~(size_t)255;
    return p;
  };
  unsigned short* wq_b = (unsigned short*)alloc((size_t)DIMD * DIMD * 2);
  unsigned short* wk_b = (unsigned short*)alloc((size_t)DIMD * DIMD * 2);
  unsigned short* wv_b = (unsigned short*)alloc((size_t)DIMD * DIMD * 2);
  unsigned short* wo_b = (unsigned short*)alloc((size_t)DIMD * DIMD * 2);
  unsigned short* w1_b = (unsigned short*)alloc((size_t)DIMD * FFDIM * 2);
  unsigned short* w2_b = (unsigned short*)alloc((size_t)FFDIM * DIMD * 2);
  unsigned short* n1   = (unsigned short*)alloc((size_t)M * DIMD * 2);
  unsigned short* Qb   = (unsigned short*)alloc((size_t)M * DIMD * 2);
  unsigned short* Kb   = (unsigned short*)alloc((size_t)M * DIMD * 2);
  unsigned short* Vb   = (unsigned short*)alloc((size_t)M * DIMD * 2);
  unsigned short* attn = (unsigned short*)alloc((size_t)M * DIMD * 2);
  float*          x2   = (float*)         alloc((size_t)M * DIMD * 4);
  unsigned short* h2   = (unsigned short*)alloc((size_t)M * DIMD * 2);
  unsigned short* a1   = (unsigned short*)alloc((size_t)M * FFDIM * 2);

  dim3 blk(256);
  k_f2bf<<<1024, blk, 0, stream>>>(wq, wq_b, DIMD * DIMD);
  k_f2bf<<<1024, blk, 0, stream>>>(wk, wk_b, DIMD * DIMD);
  k_f2bf<<<1024, blk, 0, stream>>>(wv, wv_b, DIMD * DIMD);
  k_f2bf<<<1024, blk, 0, stream>>>(wo, wo_b, DIMD * DIMD);
  k_f2bf<<<2048, blk, 0, stream>>>(w1, w1_b, DIMD * FFDIM);
  k_f2bf<<<2048, blk, 0, stream>>>(w2, w2_b, FFDIM * DIMD);

  k_layernorm<<<M, blk, 0, stream>>>(x, g1, be1, n1);

  dim3 gD(DIMD / 128, M / 128);              // (8, 32)
  k_gemm_bf16<<<gD, blk, 0, stream>>>(n1, wq_b, bq, nullptr, nullptr, Qb, M, DIMD, DIMD, 0);
  k_gemm_bf16<<<gD, blk, 0, stream>>>(n1, wk_b, bk, nullptr, nullptr, Kb, M, DIMD, DIMD, 0);
  k_gemm_bf16<<<gD, blk, 0, stream>>>(n1, wv_b, bv, nullptr, nullptr, Vb, M, DIMD, DIMD, 0);

  dim3 gA(SEQ / 128, BATCH * NHEAD);         // (16, 32)
  k_attention<<<gA, blk, 0, stream>>>(Qb, Kb, Vb, attn);

  k_gemm_bf16<<<gD, blk, 0, stream>>>(attn, wo_b, bo, x, x2, nullptr, M, DIMD, DIMD, 0);

  k_layernorm<<<M, blk, 0, stream>>>(x2, g2, be2, h2);

  dim3 gF(FFDIM / 128, M / 128);             // (32, 32)
  k_gemm_bf16<<<gF, blk, 0, stream>>>(h2, w1_b, b1, nullptr, nullptr, a1, M, FFDIM, DIMD, 1);
  k_gemm_bf16<<<gD, blk, 0, stream>>>(a1, w2_b, b2, x2, (float*)d_out, nullptr, M, DIMD, FFDIM, 0);
}
